// RoIProcessingNetwork_66503273611610
// MI455X (gfx1250) — matrix operand order repaired and checked by reading, not compiled, not run on hardware
//
#include <hip/hip_runtime.h>
#include <hip/hip_bf16.h>

// ---------------------------------------------------------------------------
// CDNA5 (gfx1250, wave32) RoI head pipeline:
//   1) roi_align_kernel : proposals -> roi features (2048 x 12544) in bf16
//   2) gemm_bf16_wmma   : FC6  (K=12544) -> relu -> h6 bf16 (2048 x 1024)
//   3) gemm_bf16_wmma   : FC7  (K=1024)  -> relu -> h7 f32  (2048 x 1024)
//   4) head_decode      : cls/reg dot products + softmax + box decode
//   5) nms_kernel       : per-image top-100 NMS, writes all 4 outputs
//
// GEMM: V_WMMA_F32_16X16X32_BF16, 2x2 register blocking per wave (32x32 out),
// block tile 64x128, double-buffered LDS with GLOBAL_LOAD_ASYNC_TO_LDS_B128
// issued a stage ahead so the DMA overlaps the WMMA pipe (ASYNCcnt).
// ---------------------------------------------------------------------------

typedef __bf16 bf16_t;
typedef bf16_t v16bf __attribute__((ext_vector_type(16)));
typedef float  v8f   __attribute__((ext_vector_type(8)));

#define N_PROP   2048
#define CMID     1024
#define ROIK     12544   // 256 channels * 49 bins
#define NIMG     8
#define PERIM    256
#define IMTOP    100

// ---------------------------------------------------------------------------
// 1) RoI align: one block per proposal, one thread per channel (256).
// ---------------------------------------------------------------------------
__global__ __launch_bounds__(256)
void roi_align_kernel(const float* __restrict__ proposals,
                      const int*   __restrict__ imidx,
                      const float* __restrict__ f0, const float* __restrict__ f1,
                      const float* __restrict__ f2, const float* __restrict__ f3,
                      bf16_t* __restrict__ roiA)
{
    const int p = blockIdx.x;
    const int c = threadIdx.x;

    const float x1 = proposals[p * 4 + 0];
    const float y1 = proposals[p * 4 + 1];
    const float x2 = proposals[p * 4 + 2];
    const float y2 = proposals[p * 4 + 3];

    const float area = fmaxf(x2 - x1, 0.f) * fmaxf(y2 - y1, 0.f);
    int lvl = (int)floorf(4.0f + log2f(sqrtf(area) / 224.0f + 1e-8f));
    lvl = min(5, max(2, lvl)) - 2;            // 0..3
    const int stride = 4 << lvl;
    const int W = 640 / stride;
    const int H = W;
    const float* f = (lvl == 0) ? f0 : (lvl == 1) ? f1 : (lvl == 2) ? f2 : f3;
    const int b = imidx[p];
    const float sc = 1.0f / (float)stride;

    const float* fc = f + ((size_t)(b * 256 + c) * (size_t)H) * (size_t)W;

    const float bwx = (x2 - x1) * sc * (1.0f / 7.0f);
    const float bwy = (y2 - y1) * sc * (1.0f / 7.0f);
    const float ox = x1 * sc, oy = y1 * sc;

    bf16_t* dst = roiA + (size_t)p * ROIK + (size_t)c * 49;

    for (int bin = 0; bin < 49; ++bin) {
        const int byi = bin / 7, bxi = bin - byi * 7;
        float gx = ox + ((float)bxi + 0.5f) * bwx;
        float gy = oy + ((float)byi + 0.5f) * bwy;
        gx = fminf(fmaxf(gx, 0.f), (float)(W - 1));
        gy = fminf(fmaxf(gy, 0.f), (float)(H - 1));
        const float x0f = floorf(gx), y0f = floorf(gy);
        const int x0 = (int)x0f, y0 = (int)y0f;
        const int x1i = min(x0 + 1, W - 1);
        const int y1i = min(y0 + 1, H - 1);
        const float fx = gx - x0f, fy = gy - y0f;

        const float v00 = fc[y0 * W + x0];
        const float v01 = fc[y0 * W + x1i];
        const float v10 = fc[y1i * W + x0];
        const float v11 = fc[y1i * W + x1i];
        const float v = v00 * (1.f - fy) * (1.f - fx) + v01 * (1.f - fy) * fx +
                        v10 * fy * (1.f - fx) + v11 * fy * fx;
        dst[bin] = (bf16_t)v;
    }
}

// ---------------------------------------------------------------------------
// 2/3) Tiled bf16 WMMA GEMM:  C[M,N] = relu( A[M,K] * W[N,K]^T + bias )
//   Block: 256 threads = 8 waves. Block tile 64(M) x 128(N), KC = 64.
//   Waves: 2(M) x 4(N); each wave computes a 32x32 output = 2x2 WMMA tiles.
//   Double-buffered LDS; A staged by global_load_async_to_lds_b128 issued a
//   stage ahead (overlaps WMMA); B converted f32->bf16 in VGPRs a stage ahead
//   and committed to LDS after compute. B is stored [n][k] so fragments are
//   contiguous ds_load_b128 pairs.
// ---------------------------------------------------------------------------
__global__ __launch_bounds__(256)
void gemm_bf16_wmma(const bf16_t* __restrict__ A,
                    const float*  __restrict__ W,
                    const float*  __restrict__ bias,
                    float*        __restrict__ outF,
                    bf16_t*       __restrict__ outB,
                    int M, int N, int K, int relu)
{
    __shared__ __align__(16) bf16_t As [2][64][72];    // [buf][m][k]
    __shared__ __align__(16) bf16_t Bsw[2][128][72];   // [buf][n][k]

    const int tid  = threadIdx.x;
    const int lane = tid & 31;
    const int wv   = tid >> 5;          // 0..7
    const int wM   = wv >> 2;           // 0..1  -> M offset wM*32
    const int wN   = wv & 3;            // 0..3  -> N offset wN*32
    const int hi   = lane >> 4;         // lane half (CDNA5 WMMA layout)
    const int ln   = lane & 15;

    const int m0 = blockIdx.y * 64;
    const int n0 = blockIdx.x * 128;

    // A-tile staging: thread -> (row 0..63, 16 contiguous bf16 = 32B)
    const int arow = tid >> 2;
    const int acol = (tid & 3) * 16;
    // B-tile staging: thread -> (n 0..127, 32 contiguous K f32 -> bf16, 64B)
    const int bn  = tid >> 1;
    const int bk0 = (tid & 1) * 32;

    const unsigned lds_a0 = (unsigned)(size_t)&As[0][arow][acol];
    const unsigned lds_a1 = (unsigned)(size_t)&As[1][arow][acol];
    const bf16_t* agp = A + (size_t)(m0 + arow) * K + acol;   // + k0 later
    const float*  wgp = W + (size_t)(n0 + bn) * K + bk0;      // + k0 later

    v8f acc[2][2];
    #pragma unroll
    for (int i = 0; i < 2; ++i)
        #pragma unroll
        for (int j = 0; j < 2; ++j)
            acc[i][j] = (v8f){0.f, 0.f, 0.f, 0.f, 0.f, 0.f, 0.f, 0.f};

    // ---- async A issue: two b128s, INST_OFFSET applies to LDS & global
    auto issueA = [&](unsigned lds, int k0) {
        const unsigned long long ga = (unsigned long long)(const void*)(agp + k0);
        asm volatile("global_load_async_to_lds_b128 %0, %1, off\n\t"
                     "global_load_async_to_lds_b128 %0, %1, off offset:16"
                     :: "v"(lds), "v"(ga)
                     : "memory");
    };
    // ---- load + convert one B stage into VGPRs
    union Bstage { uint4 q[4]; bf16_t h[32]; };
    auto loadB = [&](int k0, Bstage& s) {
        const float* wp = wgp + k0;
        #pragma unroll
        for (int j = 0; j < 32; ++j) s.h[j] = (bf16_t)wp[j];
    };
    auto commitB = [&](int buf, const Bstage& s) {
        #pragma unroll
        for (int q = 0; q < 4; ++q)
            *reinterpret_cast<uint4*>(&Bsw[buf][bn][bk0 + 8 * q]) = s.q[q];
    };
    // ---- one KC=64 stage of WMMA work from LDS buffer `buf`
    auto compute = [&](int buf) {
        #pragma unroll
        for (int kb = 0; kb < 64; kb += 32) {
            union { uint4 q[2]; v16bf v; } af[2], bf[2];
            #pragma unroll
            for (int i = 0; i < 2; ++i) {
                const bf16_t* ap = &As[buf][wM * 32 + i * 16 + ln][kb + 8 * hi];
                af[i].q[0] = *reinterpret_cast<const uint4*>(ap);
                af[i].q[1] = *reinterpret_cast<const uint4*>(ap + 16);
            }
            #pragma unroll
            for (int j = 0; j < 2; ++j) {
                const bf16_t* bp = &Bsw[buf][wN * 32 + j * 16 + ln][kb + 8 * hi];
                bf[j].q[0] = *reinterpret_cast<const uint4*>(bp);
                bf[j].q[1] = *reinterpret_cast<const uint4*>(bp + 16);
            }
            #pragma unroll
            for (int i = 0; i < 2; ++i)
                #pragma unroll
                for (int j = 0; j < 2; ++j)
                    acc[i][j] = __builtin_amdgcn_wmma_f32_16x16x32_bf16(
                        false, af[i].v, false, bf[j].v, (short)0, acc[i][j], false, false);
        }
    };

    // ---- prologue: fill buffer 0
    issueA(lds_a0, 0);
    {
        Bstage s0;
        loadB(0, s0);
        commitB(0, s0);
    }
    asm volatile("s_wait_asynccnt 0x0" ::: "memory");
    __syncthreads();

    // ---- main pipeline
    for (int k0 = 0; k0 < K; k0 += 64) {
        const int cur  = (k0 >> 6) & 1;
        const int more = (k0 + 64 < K);
        Bstage snxt;
        if (more) {
            issueA(cur ? lds_a0 : lds_a1, k0 + 64);   // DMA overlaps WMMA below
            loadB(k0 + 64, snxt);
        }
        compute(cur);
        if (more) {
            commitB(cur ^ 1, snxt);
            asm volatile("s_wait_asynccnt 0x0" ::: "memory");
            __syncthreads();
        }
    }

    // ---- epilogue: bias + relu, C/D layout M = r + 8*hi, N = ln
    #pragma unroll
    for (int i = 0; i < 2; ++i) {
        #pragma unroll
        for (int j = 0; j < 2; ++j) {
            #pragma unroll
            for (int r = 0; r < 8; ++r) {
                const int gr = m0 + wM * 32 + i * 16 + hi * 8 + r;
                const int gc = n0 + wN * 32 + j * 16 + ln;
                float v = acc[i][j][r] + bias[gc];
                if (relu) v = fmaxf(v, 0.f);
                if (outF) outF[(size_t)gr * N + gc] = v;
                if (outB) outB[(size_t)gr * N + gc] = (bf16_t)v;
            }
        }
    }
}

// ---------------------------------------------------------------------------
// 4) Head + box decode: one wave per proposal row (8 rows / block).
// ---------------------------------------------------------------------------
__global__ __launch_bounds__(256)
void head_decode(const float* __restrict__ h7,
                 const float* __restrict__ cls_w, const float* __restrict__ cls_b,
                 const float* __restrict__ reg_w, const float* __restrict__ reg_b,
                 const float* __restrict__ proposals,
                 const int*   __restrict__ imidx,
                 const int*   __restrict__ imsizes,
                 float* __restrict__ boxes_out,
                 float* __restrict__ scores_out)
{
    const int lane = threadIdx.x & 31;
    const int wv   = threadIdx.x >> 5;
    const int row  = blockIdx.x * 8 + wv;
    if (row >= N_PROP) return;

    const float* h = h7 + (size_t)row * CMID;
    float acc[10];
    #pragma unroll
    for (int j = 0; j < 10; ++j) acc[j] = 0.f;

    for (int k = lane; k < CMID; k += 32) {
        const float hv = h[k];
        acc[0] += hv * cls_w[k];
        acc[1] += hv * cls_w[CMID + k];
        #pragma unroll
        for (int j = 0; j < 8; ++j)
            acc[2 + j] += hv * reg_w[j * CMID + k];
    }
    #pragma unroll
    for (int j = 0; j < 10; ++j) {
        #pragma unroll
        for (int off = 16; off >= 1; off >>= 1)
            acc[j] += __shfl_xor(acc[j], off, 32);
    }

    if (lane == 0) {
        const float l0 = acc[0] + cls_b[0];
        const float l1 = acc[1] + cls_b[1];
        const float mx = fmaxf(l0, l1);
        const float e0 = __expf(l0 - mx), e1 = __expf(l1 - mx);
        const float scr = e1 / (e0 + e1);

        const float dx = (acc[2 + 4] + reg_b[4]) * 0.1f;
        const float dy = (acc[2 + 5] + reg_b[5]) * 0.1f;
        const float dw = (acc[2 + 6] + reg_b[6]) * 0.2f;
        const float dh = (acc[2 + 7] + reg_b[7]) * 0.2f;

        const float x1 = proposals[row * 4 + 0];
        const float y1 = proposals[row * 4 + 1];
        const float x2 = proposals[row * 4 + 2];
        const float y2 = proposals[row * 4 + 3];
        const float w  = x2 - x1, hh = y2 - y1;
        const float cx = x1 + 0.5f * w, cy = y1 + 0.5f * hh;
        const float pcx = dx * w + cx, pcy = dy * hh + cy;
        const float CL = 4.135166556742356f;   // log(1000/16)
        const float pw = w  * __expf(fminf(dw, CL));
        const float ph = hh * __expf(fminf(dh, CL));

        const int img = imidx[row];
        const float szw = (float)imsizes[img * 2 + 1];
        const float szh = (float)imsizes[img * 2 + 0];
        const float bx1 = fminf(fmaxf(pcx - 0.5f * pw, 0.f), szw);
        const float bx2 = fminf(fmaxf(pcx + 0.5f * pw, 0.f), szw);
        const float by1 = fminf(fmaxf(pcy - 0.5f * ph, 0.f), szh);
        const float by2 = fminf(fmaxf(pcy + 0.5f * ph, 0.f), szh);

        const bool keep = (scr > 0.05f) && (bx2 - bx1 >= 1.0f) && (by2 - by1 >= 1.0f);
        boxes_out[row * 4 + 0] = bx1;
        boxes_out[row * 4 + 1] = by1;
        boxes_out[row * 4 + 2] = bx2;
        boxes_out[row * 4 + 3] = by2;
        scores_out[row] = keep ? scr : -1.0f;
    }
}

// ---------------------------------------------------------------------------
// 5) Per-image NMS: one block per image, 100 rounds of argmax + suppression.
//    d_out layout (flat): boxes f32[8*100*4] | scr f32[8*100] |
//                         cls i32[8*100]    | valid i32[8*100]
// ---------------------------------------------------------------------------
__global__ __launch_bounds__(256)
void nms_kernel(const float* __restrict__ boxes_in,
                const float* __restrict__ scores_in,
                float* __restrict__ out)
{
    __shared__ float sbx[PERIM][4];
    __shared__ float ss[PERIM];
    __shared__ float sa[PERIM];
    __shared__ float rs[PERIM];
    __shared__ int   ri[PERIM];

    const int img = blockIdx.x;
    const int t   = threadIdx.x;
    const int gid = img * PERIM + t;

    const float b0 = boxes_in[gid * 4 + 0];
    const float b1 = boxes_in[gid * 4 + 1];
    const float b2 = boxes_in[gid * 4 + 2];
    const float b3 = boxes_in[gid * 4 + 3];
    sbx[t][0] = b0; sbx[t][1] = b1; sbx[t][2] = b2; sbx[t][3] = b3;
    ss[t] = scores_in[gid];
    sa[t] = fmaxf(b2 - b0, 0.f) * fmaxf(b3 - b1, 0.f);
    __syncthreads();

    const float NEG_INF = -__builtin_inff();

    for (int it = 0; it < IMTOP; ++it) {
        rs[t] = ss[t];
        ri[t] = t;
        __syncthreads();
        #pragma unroll
        for (int off = 128; off >= 1; off >>= 1) {
            if (t < off) {
                const float s2 = rs[t + off];
                const int   i2 = ri[t + off];
                if (s2 > rs[t] || (s2 == rs[t] && i2 < ri[t])) {
                    rs[t] = s2; ri[t] = i2;
                }
            }
            __syncthreads();
        }
        const int   i   = ri[0];
        const float sel = rs[0];

        const float xx1 = fmaxf(sbx[i][0], sbx[t][0]);
        const float yy1 = fmaxf(sbx[i][1], sbx[t][1]);
        const float xx2 = fminf(sbx[i][2], sbx[t][2]);
        const float yy2 = fminf(sbx[i][3], sbx[t][3]);
        const float inter = fmaxf(xx2 - xx1, 0.f) * fmaxf(yy2 - yy1, 0.f);
        const float iou = inter / (sa[i] + sa[t] - inter + 1e-6f);

        if (iou > 0.5f) ss[t] = NEG_INF;
        if (t == i)     ss[t] = NEG_INF;

        if (t == 0) {
            const int base = img * IMTOP + it;
            out[base * 4 + 0] = sbx[i][0];
            out[base * 4 + 1] = sbx[i][1];
            out[base * 4 + 2] = sbx[i][2];
            out[base * 4 + 3] = sbx[i][3];
            out[NIMG * IMTOP * 4 + base] = fmaxf(sel, -1.0f);
            const int valid = (sel > 0.05f) ? 1 : 0;
            ((int*)out)[NIMG * IMTOP * 5 + base] = valid;   // out_cls
            ((int*)out)[NIMG * IMTOP * 6 + base] = valid;   // valid
        }
        __syncthreads();
    }
}

// ---------------------------------------------------------------------------
extern "C" void kernel_launch(void* const* d_in, const int* in_sizes, int n_in,
                              void* d_out, int out_size, void* d_ws, size_t ws_size,
                              hipStream_t stream) {
    (void)in_sizes; (void)n_in; (void)out_size; (void)ws_size;

    const float* proposals = (const float*)d_in[0];
    const int*   imidx     = (const int*)  d_in[1];
    const float* f0        = (const float*)d_in[2];
    const float* f1        = (const float*)d_in[3];
    const float* f2        = (const float*)d_in[4];
    const float* f3        = (const float*)d_in[5];
    const int*   imsizes   = (const int*)  d_in[6];
    const float* fc6_w     = (const float*)d_in[7];
    const float* fc6_b     = (const float*)d_in[8];
    const float* fc7_w     = (const float*)d_in[9];
    const float* fc7_b     = (const float*)d_in[10];
    const float* cls_w     = (const float*)d_in[11];
    const float* cls_b     = (const float*)d_in[12];
    const float* reg_w     = (const float*)d_in[13];
    const float* reg_b     = (const float*)d_in[14];

    // workspace carve-up
    char* ws = (char*)d_ws;
    size_t off = 0;
    bf16_t* roiA = (bf16_t*)(ws + off); off += (size_t)N_PROP * ROIK * 2;   // 51.4 MB
    bf16_t* h6   = (bf16_t*)(ws + off); off += (size_t)N_PROP * CMID * 2;   //  4 MB
    float*  h7   = (float*) (ws + off); off += (size_t)N_PROP * CMID * 4;   //  8 MB
    float*  bxs  = (float*) (ws + off); off += (size_t)N_PROP * 4 * 4;
    float*  scr  = (float*) (ws + off); off += (size_t)N_PROP * 4;

    // 1) RoI align -> bf16 features
    roi_align_kernel<<<N_PROP, 256, 0, stream>>>(proposals, imidx, f0, f1, f2, f3, roiA);

    // 2) FC6: (2048 x 12544) x (12544 x 1024) + relu -> h6 bf16
    gemm_bf16_wmma<<<dim3(CMID / 128, N_PROP / 64), 256, 0, stream>>>(
        roiA, fc6_w, fc6_b, nullptr, h6, N_PROP, CMID, ROIK, 1);

    // 3) FC7: (2048 x 1024) x (1024 x 1024) + relu -> h7 f32
    gemm_bf16_wmma<<<dim3(CMID / 128, N_PROP / 64), 256, 0, stream>>>(
        h6, fc7_w, fc7_b, h7, nullptr, N_PROP, CMID, CMID, 1);

    // 4) cls/reg heads + softmax + box decode
    head_decode<<<N_PROP / 8, 256, 0, stream>>>(
        h7, cls_w, cls_b, reg_w, reg_b, proposals, imidx, imsizes, bxs, scr);

    // 5) per-image NMS + output packing
    nms_kernel<<<NIMG, 256, 0, stream>>>(bxs, scr, (float*)d_out);
}
